// TEN_28467043238259
// MI455X (gfx1250) — matrix operand order repaired and checked by reading, not compiled, not run on hardware
//
#include <hip/hip_runtime.h>
#include <cstdint>

// ---------------------------------------------------------------------------
// TEN forward for MI455X (gfx1250, wave32, WMMA).
//
// Assumed flattened input order (top-level dict order from setup_inputs();
// nested "params" flattened like a JAX pytree: dict keys sorted, lists in
// order, each linear as {b, w}):
//   [0] pos [8000,3] f32            [1] x_feat [8000,15] f32
//   [2] embed.b [128]               [3] embed.w [15,128]
//   [4 .. 111] layers[0..2]:
//       per layer (36 leaves): mp{0_0,0_1,1_1,1_2} each {inf.b, inf.w[128,1],
//       m1.b, m1.w[259,128], m2.b, m2.w[128,128]}  (24 leaves)
//       then upd{0,1,2} each {l1.b, l1.w[128,128], l2.b, l2.w[128,128]} (12)
//   [112..115] post: l1.b, l1.w[384,128], l2.b[1], l2.w[128,1]
//   [116..127] pre{0,1,2}: l1.b, l1.w, l2.b, l2.w
//   [128..130] x_ind 0/1/2 (int32)  [131..134] adj 0_0,0_1,1_1,1_2 [2,E]
//   [135..137] x_batch 0/1/2        [138] batch_size (=64)
// ---------------------------------------------------------------------------

typedef unsigned int  u32;
typedef unsigned short u16;
typedef __bf16 bf16_t;
typedef bf16_t v16bf __attribute__((ext_vector_type(16)));
typedef float  v8f   __attribute__((ext_vector_type(8)));
typedef u32    v8u   __attribute__((ext_vector_type(8)));

__device__ __forceinline__ u16 f2bf(float f) {
  u32 u = __float_as_uint(f);
  u32 r = u + 0x7FFFu + ((u >> 16) & 1u);   // round-to-nearest-even
  return (u16)(r >> 16);
}
__device__ __forceinline__ float bf2f(u16 h) {
  return __uint_as_float(((u32)h) << 16);
}
__device__ __forceinline__ float silu_f(float v) {
  return v * (1.0f / (1.0f + __expf(-v)));
}

// ---------------------------------------------------------------------------
// Weight convert: W f32 [fin,128] -> Wt bf16 [128][kpad], zero-padded K.
// ---------------------------------------------------------------------------
__global__ void convert_weight_kernel(const float* __restrict__ w,
                                      u16* __restrict__ wt, int fin, int kpad) {
  int idx = blockIdx.x * blockDim.x + threadIdx.x;
  if (idx >= 128 * kpad) return;
  int o = idx / kpad;
  int k = idx - o * kpad;
  float v = (k < fin) ? w[k * 128 + o] : 0.0f;
  wt[o * kpad + k] = f2bf(v);
}

__global__ void fill_zero_kernel(float* __restrict__ p, int n) {
  int i = blockIdx.x * blockDim.x + threadIdx.x;
  if (i < n) p[i] = 0.0f;
}

// x (+ optional agg) -> bf16
__global__ void add_convert_kernel(const float* __restrict__ x,
                                   const float* __restrict__ agg,
                                   u16* __restrict__ t, int n) {
  int i = blockIdx.x * blockDim.x + threadIdx.x;
  if (i >= n) return;
  float v = x[i] + (agg ? agg[i] : 0.0f);
  t[i] = f2bf(v);
}

// ---------------------------------------------------------------------------
// Per-cell centroid (of pos) and diameter. k = 1,2,3.
// ---------------------------------------------------------------------------
__global__ void cell_geom_kernel(const int* __restrict__ ind,
                                 const float* __restrict__ pos,
                                 float* __restrict__ cent,
                                 float* __restrict__ diam, int n, int k) {
  int i = blockIdx.x * blockDim.x + threadIdx.x;
  if (i >= n) return;
  float px[3] = {0.f, 0.f, 0.f}, py[3] = {0.f, 0.f, 0.f}, pz[3] = {0.f, 0.f, 0.f};
  float cx = 0.f, cy = 0.f, cz = 0.f;
#pragma unroll
  for (int j = 0; j < 3; ++j) {
    if (j < k) {
      int p = ind[i * k + j];
      px[j] = pos[p * 3 + 0];
      py[j] = pos[p * 3 + 1];
      pz[j] = pos[p * 3 + 2];
      cx += px[j]; cy += py[j]; cz += pz[j];
    }
  }
  float invk = 1.0f / (float)k;
  cent[i * 3 + 0] = cx * invk;
  cent[i * 3 + 1] = cy * invk;
  cent[i * 3 + 2] = cz * invk;
  float dmax = 0.0f;
  if (k > 1) {
#pragma unroll
    for (int a = 0; a < 3; ++a) {
#pragma unroll
      for (int b = 0; b < 3; ++b) {
        if (a < k && b < k) {
          float dx = px[a] - px[b], dy = py[a] - py[b], dz = pz[a] - pz[b];
          float sq = dx * dx + dy * dy + dz * dz;
          dmax = fmaxf(dmax, sqrtf(fmaxf(sq, 1e-12f)));
        }
      }
    }
  }
  diam[i] = dmax;
}

// ---------------------------------------------------------------------------
// Embed: x[d] = mean_k(x_feat[ind]) @ W[15,128] + b.  One thread per (cell,col).
// ---------------------------------------------------------------------------
__global__ void embed_kernel(const int* __restrict__ ind,
                             const float* __restrict__ xf,
                             const float* __restrict__ w,
                             const float* __restrict__ b,
                             float* __restrict__ xo, int n, int k) {
  int idx = blockIdx.x * blockDim.x + threadIdx.x;
  if (idx >= n * 128) return;
  int i = idx >> 7;
  int c = idx & 127;
  float acc = b[c];
  float invk = 1.0f / (float)k;
#pragma unroll
  for (int f = 0; f < 15; ++f) {
    float m = 0.0f;
#pragma unroll
    for (int j = 0; j < 3; ++j) {
      if (j < k) m += xf[ind[i * k + j] * 15 + f];
    }
    acc += (m * invk) * w[f * 128 + c];
  }
  xo[idx] = acc;
}

// ---------------------------------------------------------------------------
// Gather per-edge state [E, 288] bf16: [x_s | x_r | dist, diam_s, diam_r | 0pad]
// ---------------------------------------------------------------------------
__global__ void gather_state_kernel(const float* __restrict__ xs,
                                    const float* __restrict__ xr,
                                    const int* __restrict__ adj, int E,
                                    const float* __restrict__ cs,
                                    const float* __restrict__ cr,
                                    const float* __restrict__ dsnd,
                                    const float* __restrict__ drec,
                                    u16* __restrict__ S) {
  int idx = blockIdx.x * blockDim.x + threadIdx.x;
  if (idx >= E * 288) return;
  int e = idx / 288;
  int c = idx - e * 288;
  int snd = adj[e];
  int rec = adj[E + e];
  float v;
  if (c < 128) {
    v = xs[snd * 128 + c];
  } else if (c < 256) {
    v = xr[rec * 128 + (c - 128)];
  } else if (c == 256) {
    float dx = cs[snd * 3 + 0] - cr[rec * 3 + 0];
    float dy = cs[snd * 3 + 1] - cr[rec * 3 + 1];
    float dz = cs[snd * 3 + 2] - cr[rec * 3 + 2];
    v = sqrtf(fmaxf(dx * dx + dy * dy + dz * dz, 1e-12f));
  } else if (c == 257) {
    v = dsnd[snd];
  } else if (c == 258) {
    v = drec[rec];
  } else {
    v = 0.0f;
  }
  S[idx] = f2bf(v);
}

// ---------------------------------------------------------------------------
// WMMA GEMM:  out = epilogue(A[M,Kpad]_bf16 @ Wt^T + bias)
//   Wt bf16 [128][Kpad] (transposed weights), N fixed = 128.
//   act: 0=none, 1=silu.  residual/outF/outB nullable.
// Block = 256 threads (8 waves), 128 rows x 128 cols per block.
// Full weight tile resident in LDS; A k-tiles double-buffered so global-load
// latency overlaps the WMMA issue stream; global_prefetch hints 2 tiles ahead.
// Dynamic LDS: 128*K2 u32 (weights) + 2*2048 u32 (A double buffer).
// ---------------------------------------------------------------------------
__global__ void __launch_bounds__(256)
gemm_bf16_kernel(const u16* __restrict__ A, const u16* __restrict__ Wt,
                 const float* __restrict__ bias, int M, int Kpad, int act,
                 const float* __restrict__ residual, float* __restrict__ outF,
                 u16* __restrict__ outB) {
  extern __shared__ u32 smem[];
  const int K2 = Kpad >> 1;             // u32 per row
  u32* ldsW = smem;                     // [128][K2]
  u32* ldsA0 = smem + 128 * K2;         // [128][16] double buffer 0
  u32* ldsA1 = ldsA0 + 2048;            // [128][16] double buffer 1

  const int tid = threadIdx.x;
  const int wave = tid >> 5;
  const int lane = tid & 31;
  const int half = lane >> 4;
  const int lm = lane & 15;
  const size_t rowBase = (size_t)blockIdx.x * 128;
  const u32* Ag = (const u32*)A;
  const u32* Wg = (const u32*)Wt;

  // Load the full weight tile (128 x Kpad bf16) once per block.
  const int wtot = 128 * K2;
  for (int idx = tid; idx < wtot; idx += 256) ldsW[idx] = Wg[idx];

  // Per-thread A-staging assignment: 8 u32 per k-tile.
  u32 regs[8];
  auto load_tile = [&](int kt) {
#pragma unroll
    for (int it = 0; it < 8; ++it) {
      int idx = tid + (it << 8);
      int r = idx >> 4;
      int cp = idx & 15;
      size_t gr = rowBase + (size_t)r;
      u32 av = 0u;
      if (gr < (size_t)M) av = Ag[gr * (size_t)K2 + (size_t)(kt << 4) + cp];
      regs[it] = av;
    }
  };
  auto store_tile = [&](u32* buf) {
#pragma unroll
    for (int it = 0; it < 8; ++it) buf[tid + (it << 8)] = regs[it];
  };

  load_tile(0);
  store_tile(ldsA0);
  __syncthreads();

  v8f acc[8] = {};
  const int nk = Kpad >> 5;
  for (int kt = 0; kt < nk; ++kt) {
    u32* cur = (kt & 1) ? ldsA1 : ldsA0;
    // Prefetch hint two tiles ahead (cheap, fills GL2/L0 ahead of the loads).
    if (kt + 2 < nk) {
      size_t gr = rowBase + (size_t)(tid >> 1);
      if (gr < (size_t)M) {
        __builtin_prefetch(&Ag[gr * (size_t)K2 + (size_t)((kt + 2) << 4) + ((tid & 1) << 3)], 0, 1);
      }
    }
    // Start next tile's global loads before computing on the current one.
    if (kt + 1 < nk) load_tile(kt + 1);

    // A fragment: lane holds M = lane&15; VGPR i holds K pair
    //   K0 = (i>>2)*16 + (lane>>4)*8 + (i&3)*2   (ISA 16-bit A 16x32 layout)
    union { v8u u; v16bf b; } fa;
#pragma unroll
    for (int i = 0; i < 8; ++i) {
      int aoff = ((i >> 2) << 3) + (half << 2) + (i & 3);  // K0/2
      fa.u[i] = cur[(((wave << 4) + lm) << 4) + aoff];
    }

#pragma unroll
    for (int nt = 0; nt < 8; ++nt) {
      // B fragment: lane holds N = lane&15; VGPR i holds K pair
      //   K0 = (i>>2)*16 + (lane>>4)*8 + (i&3)*2   (same as A)
      union { v8u u; v16bf b; } fb;
#pragma unroll
      for (int i = 0; i < 8; ++i) {
        int boff = ((i >> 2) << 3) + (half << 2) + (i & 3);  // K0/2
        fb.u[i] = ldsW[((nt << 4) + lm) * K2 + (kt << 4) + boff];
      }
      acc[nt] = __builtin_amdgcn_wmma_f32_16x16x32_bf16(
          false, fa.b, false, fb.b, (short)0, acc[nt], false, false);
    }

    // Store next tile into the alternate buffer (not read this iteration).
    if (kt + 1 < nk) store_tile((kt & 1) ? ldsA0 : ldsA1);
    __syncthreads();
  }

  // Epilogue. C/D layout: VGPR j -> row = 16*wave + j + 8*half, col = nt*16 + lm
#pragma unroll
  for (int nt = 0; nt < 8; ++nt) {
    int col = (nt << 4) + lm;
    float bv = bias ? bias[col] : 0.0f;
#pragma unroll
    for (int j = 0; j < 8; ++j) {
      int rloc = (wave << 4) + j + (half << 3);
      size_t gr = rowBase + (size_t)rloc;
      if (gr < (size_t)M) {
        float v = acc[nt][j] + bv;
        if (act == 1) v = silu_f(v);
        if (residual) v += residual[gr * 128 + col];
        if (outF) outF[gr * 128 + col] = v;
        if (outB) outB[gr * 128 + col] = f2bf(v);
      }
    }
  }
}

// ---------------------------------------------------------------------------
// Per-edge gate + scatter: w = sigmoid(m . winf + binf); agg[rec] += m*w.
// One wave (32 lanes) per edge; lane handles 4 channels.
// ---------------------------------------------------------------------------
__global__ void weight_scatter_kernel(const u16* __restrict__ Mm,
                                      const float* __restrict__ wv,
                                      const float* __restrict__ bv,
                                      const int* __restrict__ adj, int E,
                                      float* __restrict__ agg) {
  int gid = blockIdx.x * blockDim.x + threadIdx.x;
  int e = gid >> 5;
  int lane = gid & 31;
  if (e >= E) return;
  float m[4];
  float part = 0.0f;
#pragma unroll
  for (int j = 0; j < 4; ++j) {
    int c = (lane << 2) + j;
    m[j] = bf2f(Mm[(size_t)e * 128 + c]);
    part += m[j] * wv[c];
  }
  for (int s = 16; s > 0; s >>= 1) part += __shfl_xor(part, s, 32);
  float wgt = 1.0f / (1.0f + __expf(-(part + bv[0])));
  int rec = adj[E + e];
#pragma unroll
  for (int j = 0; j < 4; ++j) {
    atomicAdd(&agg[(size_t)rec * 128 + (lane << 2) + j], m[j] * wgt);
  }
}

// ---------------------------------------------------------------------------
// Segment-sum pooling: pooled[batch[i], coloff + c] += Q[i, c]
// ---------------------------------------------------------------------------
__global__ void pool_kernel(const float* __restrict__ Q,
                            const int* __restrict__ xb, int n,
                            float* __restrict__ pooled, int coloff) {
  int idx = blockIdx.x * blockDim.x + threadIdx.x;
  if (idx >= n * 128) return;
  int i = idx >> 7;
  int c = idx & 127;
  atomicAdd(&pooled[(size_t)xb[i] * 384 + coloff + c], Q[idx]);
}

// ---------------------------------------------------------------------------
// Final 128 -> 1 linear: out[b] = H[b,:] . w + bias.  One wave per row.
// ---------------------------------------------------------------------------
__global__ void post_final_kernel(const u16* __restrict__ Hc,
                                  const float* __restrict__ w,
                                  const float* __restrict__ b,
                                  float* __restrict__ out, int Bn) {
  int gid = blockIdx.x * blockDim.x + threadIdx.x;
  int row = gid >> 5;
  int lane = gid & 31;
  if (row >= Bn) return;
  float part = 0.0f;
#pragma unroll
  for (int j = 0; j < 4; ++j) {
    int c = (lane << 2) + j;
    part += bf2f(Hc[row * 128 + c]) * w[c];
  }
  for (int s = 16; s > 0; s >>= 1) part += __shfl_xor(part, s, 32);
  if (lane == 0) out[row] = part + b[0];
}

// ---------------------------------------------------------------------------
// Host orchestration
// ---------------------------------------------------------------------------
static inline dim3 GRID1(long long n) { return dim3((unsigned)((n + 255) / 256)); }

extern "C" void kernel_launch(void* const* d_in, const int* in_sizes, int n_in,
                              void* d_out, int out_size, void* d_ws, size_t ws_size,
                              hipStream_t stream) {
  (void)n_in; (void)out_size; (void)ws_size;
  const int H = 128;
  const int Bn = 64;

  const float* pos    = (const float*)d_in[0];
  const float* xfeat  = (const float*)d_in[1];
  const float* embed_b = (const float*)d_in[2];
  const float* embed_w = (const float*)d_in[3];

  // layer param accessors
  auto LW = [&](int L, int a, int which) -> const float* {
    // which: 0 inf_b, 1 inf_w, 2 m1_b, 3 m1_w, 4 m2_b, 5 m2_w
    return (const float*)d_in[4 + L * 36 + a * 6 + which];
  };
  auto UW = [&](int L, int d, int which) -> const float* {
    // which: 0 l1_b, 1 l1_w, 2 l2_b, 3 l2_w
    return (const float*)d_in[4 + L * 36 + 24 + d * 4 + which];
  };
  const float* post_l1_b = (const float*)d_in[112];
  const float* post_l1_w = (const float*)d_in[113];
  const float* post_l2_b = (const float*)d_in[114];
  const float* post_l2_w = (const float*)d_in[115];
  auto PW = [&](int d, int which) -> const float* {
    return (const float*)d_in[116 + d * 4 + which];
  };

  const int* xind[3]   = {(const int*)d_in[128], (const int*)d_in[129], (const int*)d_in[130]};
  const int* adjp[4]   = {(const int*)d_in[131], (const int*)d_in[132], (const int*)d_in[133], (const int*)d_in[134]};
  const int* xbatch[3] = {(const int*)d_in[135], (const int*)d_in[136], (const int*)d_in[137]};

  int Nc[3];
  Nc[0] = in_sizes[128];        // k=1
  Nc[1] = in_sizes[129] / 2;    // k=2
  Nc[2] = in_sizes[130] / 3;    // k=3
  const int Ntot = Nc[0] + Nc[1] + Nc[2];
  int Ea[4];
  int Emax = 0;
  for (int a = 0; a < 4; ++a) {
    Ea[a] = in_sizes[131 + a] / 2;
    if (Ea[a] > Emax) Emax = Ea[a];
  }
  const int adjS[4] = {0, 0, 1, 1};
  const int adjR[4] = {0, 1, 1, 2};

  // ---------------- workspace carve-out ----------------
  char* ws = (char*)d_ws;
  size_t off = 0;
  auto alloc = [&](size_t bytes) -> void* {
    void* p = (void*)(ws + off);
    off = (off + bytes + 255) & ~(size_t)255;
    return p;
  };

  float* x    = (float*)alloc((size_t)Ntot * H * 4);
  float* agg  = (float*)alloc((size_t)Ntot * H * 4);   // reused as Q after layers
  u16*   t    = (u16*)  alloc((size_t)Ntot * H * 2);
  u16*   g    = (u16*)  alloc((size_t)Ntot * H * 2);
  u16*   S    = (u16*)  alloc((size_t)Emax * 288 * 2);
  u16*   H1   = (u16*)  alloc((size_t)Emax * H * 2);
  u16*   Mm   = (u16*)  alloc((size_t)Emax * H * 2);
  float* cent = (float*)alloc((size_t)Ntot * 3 * 4);
  float* diam = (float*)alloc((size_t)Ntot * 4);
  float* pooled  = (float*)alloc((size_t)Bn * 384 * 4);
  u16*   pooledb = (u16*)  alloc((size_t)Bn * 384 * 2);
  u16*   postH   = (u16*)  alloc((size_t)Bn * H * 2);

  u16* m1t[3][4]; u16* m2t[3][4]; u16* u1t[3][3]; u16* u2t[3][3];
  u16* p1t[3]; u16* p2t[3]; u16* post1t;
  for (int L = 0; L < 3; ++L) {
    for (int a = 0; a < 4; ++a) {
      m1t[L][a] = (u16*)alloc((size_t)128 * 288 * 2);
      m2t[L][a] = (u16*)alloc((size_t)128 * 128 * 2);
    }
    for (int d = 0; d < 3; ++d) {
      u1t[L][d] = (u16*)alloc((size_t)128 * 128 * 2);
      u2t[L][d] = (u16*)alloc((size_t)128 * 128 * 2);
    }
  }
  for (int d = 0; d < 3; ++d) {
    p1t[d] = (u16*)alloc((size_t)128 * 128 * 2);
    p2t[d] = (u16*)alloc((size_t)128 * 128 * 2);
  }
  post1t = (u16*)alloc((size_t)128 * 384 * 2);

  int rowoff[3] = {0, Nc[0], Nc[0] + Nc[1]};
  float* xd[3]; float* aggd[3]; u16* td[3]; u16* gd[3];
  float* centd[3]; float* diamd[3]; float* Qd[3];
  for (int d = 0; d < 3; ++d) {
    xd[d]    = x    + (size_t)rowoff[d] * H;
    aggd[d]  = agg  + (size_t)rowoff[d] * H;
    Qd[d]    = aggd[d];                     // pre-pool output aliases agg
    td[d]    = t    + (size_t)rowoff[d] * H;
    gd[d]    = g    + (size_t)rowoff[d] * H;
    centd[d] = cent + (size_t)rowoff[d] * 3;
    diamd[d] = diam + (size_t)rowoff[d];
  }

  auto conv_w = [&](const float* w, u16* wt, int fin, int kpad) {
    long long n = (long long)128 * kpad;
    convert_weight_kernel<<<GRID1(n), 256, 0, stream>>>(w, wt, fin, kpad);
  };
  auto gemm = [&](const u16* A, const u16* Wt, const float* bias, int M, int Kpad,
                  int act, const float* resid, float* outF, u16* outB) {
    dim3 grd((M + 127) / 128);
    size_t shBytes = (size_t)256 * Kpad + 16384;  // W tile + A double buffer
    gemm_bf16_kernel<<<grd, 256, shBytes, stream>>>(A, Wt, bias, M, Kpad, act, resid, outF, outB);
  };

  // ---------------- weight conversion (transpose + pad + bf16) ----------------
  for (int L = 0; L < 3; ++L) {
    for (int a = 0; a < 4; ++a) {
      conv_w(LW(L, a, 3), m1t[L][a], 259, 288);
      conv_w(LW(L, a, 5), m2t[L][a], 128, 128);
    }
    for (int d = 0; d < 3; ++d) {
      conv_w(UW(L, d, 1), u1t[L][d], 128, 128);
      conv_w(UW(L, d, 3), u2t[L][d], 128, 128);
    }
  }
  for (int d = 0; d < 3; ++d) {
    conv_w(PW(d, 1), p1t[d], 128, 128);
    conv_w(PW(d, 3), p2t[d], 128, 128);
  }
  conv_w(post_l1_w, post1t, 384, 384);

  // ---------------- geometry + embedding ----------------
  for (int d = 0; d < 3; ++d) {
    cell_geom_kernel<<<GRID1(Nc[d]), 256, 0, stream>>>(xind[d], pos, centd[d], diamd[d], Nc[d], d + 1);
    embed_kernel<<<GRID1((long long)Nc[d] * H), 256, 0, stream>>>(xind[d], xfeat, embed_w, embed_b,
                                                                  xd[d], Nc[d], d + 1);
  }

  // ---------------- message-passing layers ----------------
  for (int L = 0; L < 3; ++L) {
    fill_zero_kernel<<<GRID1((long long)Ntot * H), 256, 0, stream>>>(agg, Ntot * H);
    for (int a = 0; a < 4; ++a) {
      int s = adjS[a], r = adjR[a], E = Ea[a];
      gather_state_kernel<<<GRID1((long long)E * 288), 256, 0, stream>>>(
          xd[s], xd[r], adjp[a], E, centd[s], centd[r], diamd[s], diamd[r], S);
      gemm(S,  m1t[L][a], LW(L, a, 2), E, 288, 1, nullptr, nullptr, H1);
      gemm(H1, m2t[L][a], LW(L, a, 4), E, 128, 1, nullptr, nullptr, Mm);
      weight_scatter_kernel<<<GRID1((long long)E * 32), 256, 0, stream>>>(
          Mm, LW(L, a, 1), LW(L, a, 0), adjp[a], E, aggd[r]);
    }
    for (int d = 0; d < 3; ++d) {
      long long n = (long long)Nc[d] * H;
      add_convert_kernel<<<GRID1(n), 256, 0, stream>>>(xd[d], aggd[d], td[d], (int)n);
      gemm(td[d], u1t[L][d], UW(L, d, 0), Nc[d], 128, 1, nullptr, nullptr, gd[d]);
      gemm(gd[d], u2t[L][d], UW(L, d, 2), Nc[d], 128, 0, xd[d], xd[d], nullptr);
    }
  }

  // ---------------- pre-pool MLP + pooling ----------------
  for (int d = 0; d < 3; ++d) {
    long long n = (long long)Nc[d] * H;
    add_convert_kernel<<<GRID1(n), 256, 0, stream>>>(xd[d], nullptr, td[d], (int)n);
    gemm(td[d], p1t[d], PW(d, 0), Nc[d], 128, 1, nullptr, nullptr, gd[d]);
    gemm(gd[d], p2t[d], PW(d, 2), Nc[d], 128, 0, nullptr, Qd[d], nullptr);
  }
  fill_zero_kernel<<<GRID1((long long)Bn * 384), 256, 0, stream>>>(pooled, Bn * 384);
  for (int d = 0; d < 3; ++d) {
    pool_kernel<<<GRID1((long long)Nc[d] * H), 256, 0, stream>>>(Qd[d], xbatch[d], Nc[d], pooled, d * H);
  }

  // ---------------- post MLP ----------------
  add_convert_kernel<<<GRID1((long long)Bn * 384), 256, 0, stream>>>(pooled, nullptr, pooledb, Bn * 384);
  gemm(pooledb, post1t, post_l1_b, Bn, 384, 1, nullptr, nullptr, postH);
  post_final_kernel<<<GRID1((long long)Bn * 32), 256, 0, stream>>>(postH, post_l2_w, post_l2_b,
                                                                   (float*)d_out, Bn);
}
